// XorNeuronMLP_v3_32315333935409
// MI455X (gfx1250) — matrix operand order repaired and checked by reading, not compiled, not run on hardware
//
#include <hip/hip_runtime.h>
#include <hip/hip_bf16.h>
#include <math.h>

// ---------------------------------------------------------------------------
// Types for CDNA5 WMMA (wave32): v_wmma_f32_16x16x32_bf16
// ---------------------------------------------------------------------------
typedef __attribute__((ext_vector_type(8)))  __bf16 v8bf;
typedef __attribute__((ext_vector_type(16))) __bf16 v16bf;
typedef __attribute__((ext_vector_type(8)))  float  v8f;

static __device__ __forceinline__ v16bf cat8(v8bf lo, v8bf hi) {
  return __builtin_shufflevector(lo, hi, 0,1,2,3,4,5,6,7,8,9,10,11,12,13,14,15);
}
static __device__ __forceinline__ unsigned short f32_to_bf16_bits(float f) {
  return __builtin_bit_cast(unsigned short, (__bf16)f);
}
static __device__ __forceinline__ float bf16_bits_to_f32(unsigned short u) {
  return (float)__builtin_bit_cast(__bf16, u);
}

// ---------------------------------------------------------------------------
// CDNA5 async Global->LDS path (ASYNCcnt-tracked), guarded for portability.
// Builtin signature (from compiler diagnostic): params are pointers to
// int __attribute__((vector_size(16))) in AS(1) (global src) / AS(3) (LDS dst),
// plus immediate offset and cpol.
// ---------------------------------------------------------------------------
#if defined(__has_builtin)
#if __has_builtin(__builtin_amdgcn_global_load_async_to_lds_b128) && \
    __has_builtin(__builtin_amdgcn_s_wait_asynccnt)
#define GEMM_ASYNC 1
#endif
#endif

#ifdef GEMM_ASYNC
typedef int v4i_t __attribute__((vector_size(16)));
typedef __attribute__((address_space(1))) v4i_t gv4i_t;
typedef __attribute__((address_space(3))) v4i_t lv4i_t;
// Flat LDS addresses carry the LDS byte offset in the low 32 bits; flat global
// addresses equal the global address. Integer casts produce the AS pointers.
static __device__ __forceinline__ lv4i_t* to_lds4(void* p) {
  return (lv4i_t*)(unsigned int)(unsigned long long)p;
}
static __device__ __forceinline__ gv4i_t* to_glb4(const void* p) {
  return (gv4i_t*)(unsigned long long)p;
}
#endif

// ---------------------------------------------------------------------------
// Cast fp32 -> bf16 (4 elements / thread)
// ---------------------------------------------------------------------------
__global__ __launch_bounds__(256)
void cast_f32_to_bf16_kernel(const float* __restrict__ src,
                             unsigned short* __restrict__ dst, int n4) {
  int i = blockIdx.x * 256 + threadIdx.x;
  if (i < n4) {
    float4 v = ((const float4*)src)[i];
    uint2 o;
    o.x = (unsigned)f32_to_bf16_bits(v.x) | ((unsigned)f32_to_bf16_bits(v.y) << 16);
    o.y = (unsigned)f32_to_bf16_bits(v.z) | ((unsigned)f32_to_bf16_bits(v.w) << 16);
    ((uint2*)dst)[i] = o;
  }
}

// ---------------------------------------------------------------------------
// GEMM: Y[M,N] = A[M,K](bf16) @ W[N,K](bf16)^T + bias[N]   (f32 accumulate)
// Block tile 128x128, K-tile 32, 256 threads = 8 waves (4 M x 2 N),
// each wave -> 32x64 patch = 2x4 WMMA accumulators.
// Double-buffered async Global->LDS pipeline (ASYNCcnt) when available.
// ---------------------------------------------------------------------------
#define BM 128
#define BN 128
#define BK 32
#define LDS_STRIDE 40   // bf16 elems per tile row (32 + 8 pad, 16B aligned)

__global__ __launch_bounds__(256)
void gemm_bf16_wmma(const unsigned short* __restrict__ A,
                    const unsigned short* __restrict__ W,
                    const float* __restrict__ bias,
                    float* __restrict__ Y,
                    int M, int N, int K) {
  __shared__ __align__(16) unsigned short sA[2][BM * LDS_STRIDE];
  __shared__ __align__(16) unsigned short sB[2][BN * LDS_STRIDE];

  const int tid  = threadIdx.x;
  const int lane = tid & 31;
  const int wave = tid >> 5;
  const int wm = wave >> 1;           // 0..3 : 32-row strip
  const int wn = wave & 1;            // 0..1 : 64-col strip
  const int m0 = blockIdx.y * BM;
  const int n0 = blockIdx.x * BN;

  // tile loading: 2 threads per row, 32B (16 bf16) each
  const int r = tid >> 1;             // 0..127
  const int h = tid & 1;
  const unsigned short* ag = A + (size_t)(m0 + r) * K + h * 16;
  const unsigned short* bg = W + (size_t)(n0 + r) * K + h * 16;
  const int nk = K / BK;

  v8f acc[2][4] = {};

  auto issue_tile = [&](int kt, int buf) {
    const unsigned short* ap = ag + kt * BK;
    const unsigned short* bp = bg + kt * BK;
    unsigned short* da = &sA[buf][r * LDS_STRIDE + h * 16];
    unsigned short* db = &sB[buf][r * LDS_STRIDE + h * 16];
#ifdef GEMM_ASYNC
    // imm offset applies to BOTH global and LDS addresses (ISA 15.18.3)
    __builtin_amdgcn_global_load_async_to_lds_b128(to_glb4(ap), to_lds4(da), 0, 0);
    __builtin_amdgcn_global_load_async_to_lds_b128(to_glb4(ap), to_lds4(da), 16, 0);
    __builtin_amdgcn_global_load_async_to_lds_b128(to_glb4(bp), to_lds4(db), 0, 0);
    __builtin_amdgcn_global_load_async_to_lds_b128(to_glb4(bp), to_lds4(db), 16, 0);
#else
    uint4 av0 = *(const uint4*)ap;
    uint4 av1 = *(const uint4*)(ap + 8);
    uint4 bv0 = *(const uint4*)bp;
    uint4 bv1 = *(const uint4*)(bp + 8);
    if (kt + 1 < nk) {                 // global_prefetch_b8 next K-tile
      __builtin_prefetch(ap + BK, 0, 1);
      __builtin_prefetch(bp + BK, 0, 1);
    }
    *(uint4*)da = av0;  *(uint4*)(da + 8) = av1;
    *(uint4*)db = bv0;  *(uint4*)(db + 8) = bv1;
#endif
  };

  auto compute_tile = [&](int buf) {
    const int rl = lane & 15;
    const int ah = lane >> 4;          // K-half selector
    v16bf afrag[2];
#pragma unroll
    for (int mf = 0; mf < 2; ++mf) {
      // A 16x32: lanes 0-15 rows, V0-3 = K[8h..8h+7], V4-7 = K[8h+16..8h+23]
      const unsigned short* p =
          &sA[buf][(wm * 32 + mf * 16 + rl) * LDS_STRIDE + ah * 8];
      v8bf lo = *(const v8bf*)p;
      v8bf hi = *(const v8bf*)(p + 16);
      afrag[mf] = cat8(lo, hi);
    }
    v16bf bfrag[4];
#pragma unroll
    for (int nf = 0; nf < 4; ++nf) {
      // B 32x16: lane n holds col n; V0-3 = K[8h..8h+7], V4-7 = K[8h+16..8h+23]
      const unsigned short* p =
          &sB[buf][(wn * 64 + nf * 16 + rl) * LDS_STRIDE + ah * 8];
      v8bf lo = *(const v8bf*)p;
      v8bf hi = *(const v8bf*)(p + 16);
      bfrag[nf] = cat8(lo, hi);
    }
#pragma unroll
    for (int mf = 0; mf < 2; ++mf)
#pragma unroll
      for (int nf = 0; nf < 4; ++nf)
        acc[mf][nf] = __builtin_amdgcn_wmma_f32_16x16x32_bf16(
            false, afrag[mf], false, bfrag[nf], (short)0, acc[mf][nf],
            false, false);
  };

#ifdef GEMM_ASYNC
  issue_tile(0, 0);
  for (int kt = 0; kt < nk; ++kt) {
    const int cur = kt & 1;
    if (kt + 1 < nk) {
      issue_tile(kt + 1, cur ^ 1);     // buffer cur^1 freed by kt-1's barrier
      __builtin_amdgcn_s_wait_asynccnt(4);   // tile kt landed (in-order)
    } else {
      __builtin_amdgcn_s_wait_asynccnt(0);
    }
    __syncthreads();                   // all waves' tile-kt data visible
    compute_tile(cur);
    __syncthreads();                   // done reading before overwrite
  }
#else
  for (int kt = 0; kt < nk; ++kt) {
    issue_tile(kt, 0);
    __syncthreads();
    compute_tile(0);
    __syncthreads();
  }
#endif

  // epilogue: C/D layout — lane 0-15: M=r, N=lane; lane 16-31: M=r+8
  const int cl   = lane & 15;
  const int radd = (lane >> 4) * 8;
#pragma unroll
  for (int mf = 0; mf < 2; ++mf) {
    const int mb = m0 + wm * 32 + mf * 16 + radd;
#pragma unroll
    for (int nf = 0; nf < 4; ++nf) {
      const int col = n0 + wn * 64 + nf * 16 + cl;
      const float bb = bias[col];
      float* yp = Y + (size_t)mb * N + col;
#pragma unroll
      for (int q = 0; q < 8; ++q)
        yp[(size_t)q * N] = acc[mf][nf][q] + bb;
    }
  }
}

// ---------------------------------------------------------------------------
// Fused LayerNorm + grouped 1x1-conv MLP (per cell type): one block per row.
// Y row [4096] -> LN -> [C=16, pt=64, A=4] -> 4->32->relu->32->relu->1
// Output bf16 [1024] per row.  Dynamic LDS: row + all conv weights (~94 KB).
// ---------------------------------------------------------------------------
__global__ __launch_bounds__(256)
void ln_conv_kernel(const float* __restrict__ Y,
                    const float* __restrict__ Wc0, const float* __restrict__ bc0,
                    const float* __restrict__ Wc1, const float* __restrict__ bc1,
                    const float* __restrict__ Wc2, const float* __restrict__ bc2,
                    unsigned short* __restrict__ outb) {
  extern __shared__ float smem[];
  float* srow = smem;               // 4096
  float* sW0  = srow + 4096;        // 2048  [16][32][4]
  float* sW1  = sW0 + 2048;         // 16384 [16][32][32]
  float* sW2  = sW1 + 16384;        // 512   [16][32]
  float* sB0  = sW2 + 512;          // 512
  float* sB1  = sB0 + 512;          // 512
  float* sB2  = sB1 + 512;          // 16
  float* sred = sB2 + 16;           // 16

  const int tid = threadIdx.x;
  const int b   = blockIdx.x;

  for (int i = tid; i < 16384; i += 256) sW1[i] = Wc1[i];
  for (int i = tid; i < 2048;  i += 256) sW0[i] = Wc0[i];
  for (int i = tid; i < 512;   i += 256) { sW2[i] = Wc2[i]; sB0[i] = bc0[i]; sB1[i] = bc1[i]; }
  if (tid < 16) sB2[tid] = bc2[tid];

  const float4* yv = (const float4*)(Y + (size_t)b * 4096);
  float s = 0.f, sq = 0.f;
#pragma unroll
  for (int j = 0; j < 4; ++j) {
    float4 v = yv[tid + 256 * j];
    ((float4*)srow)[tid + 256 * j] = v;
    s  += v.x + v.y + v.z + v.w;
    sq += v.x * v.x + v.y * v.y + v.z * v.z + v.w * v.w;
  }
#pragma unroll
  for (int off = 16; off > 0; off >>= 1) {
    s  += __shfl_down(s,  off, 32);
    sq += __shfl_down(sq, off, 32);
  }
  if ((tid & 31) == 0) { sred[tid >> 5] = s; sred[8 + (tid >> 5)] = sq; }
  __syncthreads();

  float tot = 0.f, totq = 0.f;
#pragma unroll
  for (int w = 0; w < 8; ++w) { tot += sred[w]; totq += sred[8 + w]; }
  const float mean = tot * (1.0f / 4096.0f);
  const float var  = totq * (1.0f / 4096.0f) - mean * mean;
  const float rstd = rsqrtf(var + 1e-5f);

  for (int jj = 0; jj < 4; ++jj) {
    const int q = tid + 256 * jj;     // 0..1023 : (c,n) column
    const int c = q >> 6;
    const int n = q & 63;
    const float* rp = srow + c * 256 + n * 4;
    const float in0 = (rp[0] - mean) * rstd;
    const float in1 = (rp[1] - mean) * rstd;
    const float in2 = (rp[2] - mean) * rstd;
    const float in3 = (rp[3] - mean) * rstd;

    float h1[32];
    const float* w0  = sW0 + c * 128;
    const float* bb0 = sB0 + c * 32;
#pragma unroll
    for (int o = 0; o < 32; ++o) {
      float a = bb0[o];
      a = fmaf(w0[o * 4 + 0], in0, a);
      a = fmaf(w0[o * 4 + 1], in1, a);
      a = fmaf(w0[o * 4 + 2], in2, a);
      a = fmaf(w0[o * 4 + 3], in3, a);
      h1[o] = fmaxf(a, 0.f);
    }
    const float* w1  = sW1 + c * 1024;
    const float* bb1 = sB1 + c * 32;
    const float* w2  = sW2 + c * 32;
    float out = sB2[c];
    for (int o = 0; o < 32; ++o) {
      float a = bb1[o];
      const float* wr = w1 + o * 32;
#pragma unroll
      for (int j = 0; j < 32; ++j) a = fmaf(wr[j], h1[j], a);
      out = fmaf(w2[o], fmaxf(a, 0.f), out);
    }
    outb[(size_t)b * 1024 + q] = f32_to_bf16_bits(out);
  }
}

// ---------------------------------------------------------------------------
// FC (1024 -> 100) + log-softmax NLL per-row loss term. One block per row.
// ---------------------------------------------------------------------------
__global__ __launch_bounds__(128)
void fc_loss_kernel(const unsigned short* __restrict__ actb,
                    const float* __restrict__ Wf, const float* __restrict__ bfv,
                    const int* __restrict__ labels,
                    float* __restrict__ logits, float* __restrict__ loss_terms) {
  __shared__ float sx[1024];
  __shared__ float sl[128];
  __shared__ float sr[128];
  const int b = blockIdx.x, tid = threadIdx.x;
  for (int i = tid; i < 1024; i += 128)
    sx[i] = bf16_bits_to_f32(actb[(size_t)b * 1024 + i]);
  __syncthreads();

  float lg = -3.0e38f;
  if (tid < 100) {
    const float4* w = (const float4*)(Wf + (size_t)tid * 1024);
    float a = bfv[tid];
    for (int j = 0; j < 256; ++j) {
      float4 v = w[j];
      const float* xp = sx + j * 4;
      a = fmaf(v.x, xp[0], a); a = fmaf(v.y, xp[1], a);
      a = fmaf(v.z, xp[2], a); a = fmaf(v.w, xp[3], a);
    }
    lg = a;
    logits[(size_t)b * 100 + tid] = a;
  }
  sl[tid] = lg;
  sr[tid] = lg;
  __syncthreads();
  for (int off = 64; off > 0; off >>= 1) {
    if (tid < off) sr[tid] = fmaxf(sr[tid], sr[tid + off]);
    __syncthreads();
  }
  const float m = sr[0];
  __syncthreads();
  sr[tid] = (tid < 100) ? expf(sl[tid] - m) : 0.f;
  __syncthreads();
  for (int off = 64; off > 0; off >>= 1) {
    if (tid < off) sr[tid] += sr[tid + off];
    __syncthreads();
  }
  if (tid == 0) {
    const float lse = m + logf(sr[0]);
    const int lab = labels[b];
    loss_terms[b] = lse - sl[lab];
  }
}

// Deterministic fixed-order tree reduction of 8192 loss terms -> mean.
__global__ __launch_bounds__(256)
void loss_reduce_kernel(const float* __restrict__ terms, float* __restrict__ loss) {
  __shared__ float sr[256];
  const int tid = threadIdx.x;
  float a = 0.f;
  for (int i = tid; i < 8192; i += 256) a += terms[i];
  sr[tid] = a;
  __syncthreads();
  for (int off = 128; off > 0; off >>= 1) {
    if (tid < off) sr[tid] += sr[tid + off];
    __syncthreads();
  }
  if (tid == 0) loss[0] = sr[0] * (1.0f / 8192.0f);
}

// ---------------------------------------------------------------------------
extern "C" void kernel_launch(void* const* d_in, const int* in_sizes, int n_in,
                              void* d_out, int out_size, void* d_ws, size_t ws_size,
                              hipStream_t stream) {
  const float* x    = (const float*)d_in[0];
  const int*   labels = (const int*)d_in[1];
  const float* W0   = (const float*)d_in[2];
  const float* b0   = (const float*)d_in[3];
  const float* W1   = (const float*)d_in[4];
  const float* b1   = (const float*)d_in[5];
  const float* Wc0  = (const float*)d_in[6];
  const float* bc0  = (const float*)d_in[7];
  const float* Wc1  = (const float*)d_in[8];
  const float* bc1  = (const float*)d_in[9];
  const float* Wc2  = (const float*)d_in[10];
  const float* bc2  = (const float*)d_in[11];
  const float* Wf   = (const float*)d_in[12];
  const float* bfv  = (const float*)d_in[13];

  char* ws = (char*)d_ws;
  unsigned short* xb  = (unsigned short*)(ws);                         // 16 MB
  unsigned short* w0b = (unsigned short*)(ws + (size_t)16  * 1048576); //  8 MB
  unsigned short* w1b = (unsigned short*)(ws + (size_t)24  * 1048576); //  8 MB
  float*          Yb  = (float*)         (ws + (size_t)32  * 1048576); // 128 MB
  unsigned short* a1b = (unsigned short*)(ws + (size_t)160 * 1048576); // 16 MB
  float*          terms = (float*)       (ws + (size_t)176 * 1048576); // 32 KB

  float* logits = (float*)d_out;
  float* loss   = logits + 8192 * 100;

  cast_f32_to_bf16_kernel<<<8192, 256, 0, stream>>>(x,  xb,  8192 * 1024 / 4);
  cast_f32_to_bf16_kernel<<<4096, 256, 0, stream>>>(W0, w0b, 4096 * 1024 / 4);
  cast_f32_to_bf16_kernel<<<4096, 256, 0, stream>>>(W1, w1b, 4096 * 1024 / 4);

  dim3 ggrid(4096 / BN, 8192 / BM);
  const size_t smem = (size_t)24096 * sizeof(float);

  gemm_bf16_wmma<<<ggrid, 256, 0, stream>>>(xb, w0b, b0, Yb, 8192, 4096, 1024);
  ln_conv_kernel<<<8192, 256, smem, stream>>>(Yb, Wc0, bc0, Wc1, bc1, Wc2, bc2, a1b);
  gemm_bf16_wmma<<<ggrid, 256, 0, stream>>>(a1b, w1b, b1, Yb, 8192, 4096, 1024);
  ln_conv_kernel<<<8192, 256, smem, stream>>>(Yb, Wc0, bc0, Wc1, bc1, Wc2, bc2, a1b);
  fc_loss_kernel<<<8192, 128, 0, stream>>>(a1b, Wf, bfv, labels, logits, terms);
  loss_reduce_kernel<<<1, 256, 0, stream>>>(terms, loss);
}
